// SharedClassifier_88673894793362
// MI455X (gfx1250) — matrix operand order repaired and checked by reading, not compiled, not run on hardware
//
#include <hip/hip_runtime.h>
#include <hip/hip_bf16.h>
#include <stdint.h>

typedef __attribute__((ext_vector_type(16))) __bf16 v16bf;
typedef __attribute__((ext_vector_type(8)))  __bf16 v8bf;
typedef __attribute__((ext_vector_type(8)))  float  v8f;

#define DEVINL __device__ __forceinline__

// ---------------------------------------------------------------- prep kernels

__global__ void k_cvt_bf16(const float* __restrict__ src, __bf16* __restrict__ dst, int n) {
  int i = blockIdx.x * blockDim.x + threadIdx.x;
  int stride = gridDim.x * blockDim.x;
  for (; i < n; i += stride) dst[i] = (__bf16)src[i];
}

// dst[N][K] = src[K][N]  (weight transpose to B-fragment friendly [out][in] bf16)
__global__ void k_transpose_bf16(const float* __restrict__ src, __bf16* __restrict__ dst,
                                 int K, int N) {
  int i = blockIdx.x * blockDim.x + threadIdx.x;
  int total = K * N;
  int stride = gridDim.x * blockDim.x;
  for (; i < total; i += stride) {
    int nn = i / K, kk = i - nn * K;
    dst[i] = (__bf16)src[kk * N + nn];
  }
}

// hypernet hidden layers: hh_j[b][u] = relu(sum_p xp[b][p]*Wj[p][u] + bj[u]), j=0..3
__global__ void k_hyper_hidden(const float* __restrict__ xp,
    const float* __restrict__ w0, const float* __restrict__ b0,
    const float* __restrict__ w1, const float* __restrict__ b1,
    const float* __restrict__ w2, const float* __restrict__ b2,
    const float* __restrict__ w3, const float* __restrict__ b3,
    __bf16* __restrict__ h0, __bf16* __restrict__ h1,
    __bf16* __restrict__ h2, __bf16* __restrict__ h3, int B)
{
  int i = blockIdx.x * blockDim.x + threadIdx.x;   // over B*32
  if (i >= B * 32) return;
  int b = i >> 5, u = i & 31;
  float p[8];
  #pragma unroll
  for (int k = 0; k < 8; ++k) p[k] = xp[b * 8 + k];
  const float* W[4]  = {w0, w1, w2, w3};
  const float* Bb[4] = {b0, b1, b2, b3};
  __bf16* H[4]       = {h0, h1, h2, h3};
  #pragma unroll
  for (int j = 0; j < 4; ++j) {
    float acc = Bb[j][u];
    #pragma unroll
    for (int k = 0; k < 8; ++k) acc += p[k] * W[j][k * 32 + u];
    H[j][i] = (__bf16)fmaxf(acc, 0.f);
  }
}

// ---------------------------------------------------------------- WMMA panel helper
// C[MR*16, NT*16] = A[MR*16, K] * B[K, NT*16] + bias(col). A row-major (stride
// astride elems); Bm transposed weights [col][K]. One B fragment feeds MR WMMAs.
// Fragment layouts per CDNA5 ISA 7.12.2 (16-bit A 16x32; B 32x16).
template<int MR, int NT, typename AP, typename BP>
DEVINL void mma_panel(AP A, int astride, BP Bm, int bstride,
                      const float* __restrict__ biasC,
                      int ksteps, int colbase, int lane, v8f* acc)
{
  const int hh  = lane >> 4;     // half-wave select
  const int l15 = lane & 15;
  #pragma unroll
  for (int t = 0; t < NT; ++t) {
    float c = biasC ? biasC[colbase + t * 16 + l15] : 0.f;
    #pragma unroll
    for (int g = 0; g < MR; ++g)
      #pragma unroll
      for (int r = 0; r < 8; ++r) acc[g * NT + t][r] = c;
  }
  for (int k = 0; k < ksteps; ++k) {
    const int kb = k * 32;
    v16bf af[MR];
    #pragma unroll
    for (int g = 0; g < MR; ++g) {
      // A fragment: lane<16 -> K {0..7,16..23}; lane>=16 -> K {8..15,24..31}
      const auto* pa = A + (size_t)(g * 16 + l15) * astride + kb + (hh ? 8 : 0);
      v8bf alo = *(const v8bf*)(pa);
      v8bf ahi = *(const v8bf*)(pa + 16);
      af[g] = __builtin_shufflevector(alo, ahi, 0,1,2,3,4,5,6,7,8,9,10,11,12,13,14,15);
    }
    #pragma unroll
    for (int t = 0; t < NT; ++t) {
      const int col = colbase + t * 16 + l15;
      const auto* pb = Bm + (size_t)col * bstride + kb + hh * 8;   // K {0..7,16..23} / {8..15,24..31}
      v8bf blo = *(const v8bf*)(pb);
      v8bf bhi = *(const v8bf*)(pb + 16);
      v16bf bf_ = __builtin_shufflevector(blo, bhi, 0,1,2,3,4,5,6,7,8,9,10,11,12,13,14,15);
      #pragma unroll
      for (int g = 0; g < MR; ++g)
        acc[g * NT + t] = __builtin_amdgcn_wmma_f32_16x16x32_bf16(
            false, af[g], false, bf_, (short)0, acc[g * NT + t], false, false);
    }
  }
}

// ---------------------------------------------------------------- fused main kernel
// 32 batch rows per workgroup, 8 wave32s. Entire 3-layer MLP + per-sample
// group-norm affine + log-softmax fused; intermediates live in regs + LDS.
#define MROWS 32
__launch_bounds__(256)
__global__ void k_fused(const __bf16* __restrict__ x_bf,
                        const __bf16* __restrict__ w1T, const float* __restrict__ l1_b,
                        const __bf16* __restrict__ w2T, const float* __restrict__ l2_b,
                        const __bf16* __restrict__ w3T, const float* __restrict__ l3_b,
                        const __bf16* __restrict__ hh0, const __bf16* __restrict__ hh1,
                        const __bf16* __restrict__ hh2, const __bf16* __restrict__ hh3,
                        const __bf16* __restrict__ g1wT, const float* __restrict__ g1w_b,
                        const __bf16* __restrict__ g1bT, const float* __restrict__ g1b_b,
                        const __bf16* __restrict__ g2wT, const float* __restrict__ g2w_b,
                        const __bf16* __restrict__ g2bT, const float* __restrict__ g2b_b,
                        float* __restrict__ out)
{
  __shared__ __align__(16) __bf16 s_act[MROWS * 2048]; // 128 KB activation buffer
  __shared__ float s_red[2 * MROWS];    // sum, sumsq
  __shared__ float s_stat[2 * MROWS];   // mean, rstd  (reused: max / max+lse)
  __shared__ float s_pm[MROWS * 8];     // per-wave partials (stage 3)

  const int tid  = threadIdx.x;
  const int lane = tid & 31, wave = tid >> 5;
  const int hh   = lane >> 4, l15 = lane & 15;
  const int m0   = blockIdx.x * MROWS;

  if (tid < 2 * MROWS) s_red[tid] = 0.f;
  __syncthreads();

  // ================= stage 1: x[32,512] @ l1[512,2048], GN(1 group) affine+relu
  {
    v8f acc[2 * 16];
    mma_panel<2, 16>(x_bf + (size_t)m0 * 512, 512, w1T, 512, l1_b, 16, wave * 256, lane, acc);
    #pragma unroll
    for (int g = 0; g < 2; ++g)
      #pragma unroll
      for (int r = 0; r < 8; ++r) {
        float s = 0.f, ss = 0.f;
        #pragma unroll
        for (int t = 0; t < 16; ++t) { float v = acc[g * 16 + t][r]; s += v; ss += v * v; }
        const int row = g * 16 + r + hh * 8;
        atomicAdd(&s_red[row], s);
        atomicAdd(&s_red[MROWS + row], ss);
      }
    __syncthreads();
    if (tid < MROWS) {
      float mean = s_red[tid] * (1.f / 2048.f);
      float var  = s_red[MROWS + tid] * (1.f / 2048.f) - mean * mean;
      s_stat[tid]         = mean;
      s_stat[MROWS + tid] = rsqrtf(var + 1e-5f);
      s_red[tid] = 0.f; s_red[MROWS + tid] = 0.f;   // re-zero for stage 2
    }
    __syncthreads();
    #pragma unroll 1
    for (int t = 0; t < 16; ++t) {
      const int cb = wave * 256 + t * 16;
      const int col = cb + l15;
      #pragma unroll 1
      for (int g = 0; g < 2; ++g) {
        v8f wv, bv;   // per-sample affine from hypernet: [16,32]@[32,16] WMMA each
        mma_panel<1, 1>(hh0 + (size_t)(m0 + g * 16) * 32, 32, g1wT, 32, g1w_b, 1, cb, lane, &wv);
        mma_panel<1, 1>(hh1 + (size_t)(m0 + g * 16) * 32, 32, g1bT, 32, g1b_b, 1, cb, lane, &bv);
        #pragma unroll
        for (int r = 0; r < 8; ++r) {
          const int row = g * 16 + r + hh * 8;
          float xn = (acc[g * 16 + t][r] - s_stat[row]) * s_stat[MROWS + row];
          float y  = fmaxf(xn * wv[r] + bv[r], 0.f);
          s_act[row * 2048 + col] = (__bf16)y;
        }
      }
    }
  }
  __syncthreads();

  // ================= stage 2: act1[32,2048] @ l2[2048,1024], GN affine+relu
  {
    v8f acc[2 * 8];
    mma_panel<2, 8>((const __bf16*)s_act, 2048, w2T, 2048, l2_b, 64, wave * 128, lane, acc);
    #pragma unroll
    for (int g = 0; g < 2; ++g)
      #pragma unroll
      for (int r = 0; r < 8; ++r) {
        float s = 0.f, ss = 0.f;
        #pragma unroll
        for (int t = 0; t < 8; ++t) { float v = acc[g * 8 + t][r]; s += v; ss += v * v; }
        const int row = g * 16 + r + hh * 8;
        atomicAdd(&s_red[row], s);
        atomicAdd(&s_red[MROWS + row], ss);
      }
    __syncthreads();
    if (tid < MROWS) {
      float mean = s_red[tid] * (1.f / 1024.f);
      float var  = s_red[MROWS + tid] * (1.f / 1024.f) - mean * mean;
      s_stat[tid]         = mean;
      s_stat[MROWS + tid] = rsqrtf(var + 1e-5f);
    }
    __syncthreads();
    #pragma unroll 1
    for (int t = 0; t < 8; ++t) {
      const int cb = wave * 128 + t * 16;
      const int col = cb + l15;
      #pragma unroll 1
      for (int g = 0; g < 2; ++g) {
        v8f wv, bv;
        mma_panel<1, 1>(hh2 + (size_t)(m0 + g * 16) * 32, 32, g2wT, 32, g2w_b, 1, cb, lane, &wv);
        mma_panel<1, 1>(hh3 + (size_t)(m0 + g * 16) * 32, 32, g2bT, 32, g2b_b, 1, cb, lane, &bv);
        #pragma unroll
        for (int r = 0; r < 8; ++r) {
          const int row = g * 16 + r + hh * 8;
          float xn = (acc[g * 8 + t][r] - s_stat[row]) * s_stat[MROWS + row];
          float y  = fmaxf(xn * wv[r] + bv[r], 0.f);
          s_act[row * 1024 + col] = (__bf16)y;   // stage-3 A buffer (all reads done)
        }
      }
    }
  }
  __syncthreads();

  // ================= stage 3: act2[32,1024] @ l3[1024,256], log_softmax
  {
    v8f acc[2 * 2];
    mma_panel<2, 2>((const __bf16*)s_act, 1024, w3T, 1024, l3_b, 32, wave * 32, lane, acc);

    // row max: shuffle within 16-lane half, then cross-wave via LDS
    #pragma unroll
    for (int g = 0; g < 2; ++g)
      #pragma unroll
      for (int r = 0; r < 8; ++r) {
        float m = fmaxf(acc[g * 2 + 0][r], acc[g * 2 + 1][r]);
        #pragma unroll
        for (int off = 1; off < 16; off <<= 1) m = fmaxf(m, __shfl_xor(m, off, 32));
        if (l15 == 0) s_pm[(g * 16 + r + hh * 8) * 8 + wave] = m;
      }
    __syncthreads();
    if (tid < MROWS) {
      float m = s_pm[tid * 8];
      for (int w = 1; w < 8; ++w) m = fmaxf(m, s_pm[tid * 8 + w]);
      s_stat[tid] = m;
    }
    __syncthreads();
    #pragma unroll
    for (int g = 0; g < 2; ++g)
      #pragma unroll
      for (int r = 0; r < 8; ++r) {
        const int row = g * 16 + r + hh * 8;
        const float mx = s_stat[row];
        float e = __expf(acc[g * 2 + 0][r] - mx) + __expf(acc[g * 2 + 1][r] - mx);
        #pragma unroll
        for (int off = 1; off < 16; off <<= 1) e += __shfl_xor(e, off, 32);
        if (l15 == 0) s_pm[row * 8 + wave] = e;
      }
    __syncthreads();
    if (tid < MROWS) {
      float s = 0.f;
      for (int w = 0; w < 8; ++w) s += s_pm[tid * 8 + w];
      s_stat[MROWS + tid] = s_stat[tid] + __logf(s);
    }
    __syncthreads();
    #pragma unroll
    for (int g = 0; g < 2; ++g)
      #pragma unroll
      for (int t = 0; t < 2; ++t) {
        const int col = wave * 32 + t * 16 + l15;
        #pragma unroll
        for (int r = 0; r < 8; ++r) {
          const int row = g * 16 + r + hh * 8;
          out[(size_t)(m0 + row) * 256 + col] = acc[g * 2 + t][r] - s_stat[MROWS + row];
        }
      }
  }
}

// ---------------------------------------------------------------- launch

extern "C" void kernel_launch(void* const* d_in, const int* in_sizes, int n_in,
                              void* d_out, int out_size, void* d_ws, size_t ws_size,
                              hipStream_t stream)
{
  const float* x_feature = (const float*)d_in[0];
  const float* x_param   = (const float*)d_in[1];
  const float* hw1_1_w = (const float*)d_in[2];
  const float* hw1_1_b = (const float*)d_in[3];
  const float* hw1_2_w = (const float*)d_in[4];
  const float* hw1_2_b = (const float*)d_in[5];
  const float* hb1_1_w = (const float*)d_in[6];
  const float* hb1_1_b = (const float*)d_in[7];
  const float* hb1_2_w = (const float*)d_in[8];
  const float* hb1_2_b = (const float*)d_in[9];
  const float* hw2_1_w = (const float*)d_in[10];
  const float* hw2_1_b = (const float*)d_in[11];
  const float* hw2_2_w = (const float*)d_in[12];
  const float* hw2_2_b = (const float*)d_in[13];
  const float* hb2_1_w = (const float*)d_in[14];
  const float* hb2_1_b = (const float*)d_in[15];
  const float* hb2_2_w = (const float*)d_in[16];
  const float* hb2_2_b = (const float*)d_in[17];
  const float* l1_w = (const float*)d_in[18];
  const float* l1_b = (const float*)d_in[19];
  const float* l2_w = (const float*)d_in[20];
  const float* l2_b = (const float*)d_in[21];
  const float* l3_w = (const float*)d_in[22];
  const float* l3_b = (const float*)d_in[23];

  const int B = 32768;
  char* ws = (char*)d_ws;
  size_t off = 0;
  auto alloc = [&](size_t bytes) -> void* {
    void* p = ws + off;
    off += (bytes + 255) & ~(size_t)255;
    return p;
  };
  __bf16* x_bf = (__bf16*)alloc((size_t)B * 512 * 2);  // 32 MB
  __bf16* hh0  = (__bf16*)alloc((size_t)B * 32 * 2);
  __bf16* hh1  = (__bf16*)alloc((size_t)B * 32 * 2);
  __bf16* hh2  = (__bf16*)alloc((size_t)B * 32 * 2);
  __bf16* hh3  = (__bf16*)alloc((size_t)B * 32 * 2);
  __bf16* w1T  = (__bf16*)alloc((size_t)2048 * 512 * 2);
  __bf16* w2T  = (__bf16*)alloc((size_t)1024 * 2048 * 2);
  __bf16* w3T  = (__bf16*)alloc((size_t)256 * 1024 * 2);
  __bf16* g1wT = (__bf16*)alloc((size_t)2048 * 32 * 2);
  __bf16* g1bT = (__bf16*)alloc((size_t)2048 * 32 * 2);
  __bf16* g2wT = (__bf16*)alloc((size_t)1024 * 32 * 2);
  __bf16* g2bT = (__bf16*)alloc((size_t)1024 * 32 * 2);

  k_cvt_bf16<<<4096, 256, 0, stream>>>(x_feature, x_bf, B * 512);
  k_transpose_bf16<<<2048, 256, 0, stream>>>(l1_w, w1T, 512, 2048);
  k_transpose_bf16<<<4096, 256, 0, stream>>>(l2_w, w2T, 2048, 1024);
  k_transpose_bf16<<<512, 256, 0, stream>>>(l3_w, w3T, 1024, 256);
  k_transpose_bf16<<<256, 256, 0, stream>>>(hw1_2_w, g1wT, 32, 2048);
  k_transpose_bf16<<<256, 256, 0, stream>>>(hb1_2_w, g1bT, 32, 2048);
  k_transpose_bf16<<<128, 256, 0, stream>>>(hw2_2_w, g2wT, 32, 1024);
  k_transpose_bf16<<<128, 256, 0, stream>>>(hb2_2_w, g2bT, 32, 1024);
  k_hyper_hidden<<<(B * 32 + 255) / 256, 256, 0, stream>>>(
      x_param, hw1_1_w, hw1_1_b, hb1_1_w, hb1_1_b,
      hw2_1_w, hw2_1_b, hb2_1_w, hb2_1_b,
      hh0, hh1, hh2, hh3, B);

  k_fused<<<B / MROWS, 256, 0, stream>>>(
      x_bf, w1T, l1_b, w2T, l2_b, w3T, l3_b,
      hh0, hh1, hh2, hh3,
      g1wT, hw1_2_b, g1bT, hb1_2_b,
      g2wT, hw2_2_b, g2bT, hb2_2_b,
      (float*)d_out);
}
